// PolicyNet_85418309582992
// MI455X (gfx1250) — compile-verified, matrix operand order checked by reading, not run on hardware
//
#include <hip/hip_runtime.h>
#include <cstddef>
#include <cstdint>

// ---------------------------------------------------------------- types
typedef __bf16 bf16_t;
typedef __attribute__((ext_vector_type(16))) __bf16 v16bf;
typedef __attribute__((ext_vector_type(8)))  __bf16 v8bf;
typedef __attribute__((ext_vector_type(8)))  float  v8f;
typedef __attribute__((ext_vector_type(4)))  int    v4i;

#define TT 32
#define BB 32
#define TB (TT*BB)          // 1024 images / rows
#define HID 1024
#define NG  4096            // 4*HID gates
#define FLATK 3872          // 32*11*11

// ---------------------------------------------------------------- async LDS staging (CDNA5 path when available)
#if __has_builtin(__builtin_amdgcn_global_load_async_to_lds_b128) && \
    __has_builtin(__builtin_amdgcn_s_wait_asynccnt)
#define USE_ASYNC_LDS 1
#endif

__device__ __forceinline__ void copy16(const bf16_t* g, bf16_t* l) {
#ifdef USE_ASYNC_LDS
  __builtin_amdgcn_global_load_async_to_lds_b128(
      (__attribute__((address_space(1))) v4i*)(g),
      (__attribute__((address_space(3))) v4i*)(l), 0, 0);
#else
  *reinterpret_cast<v8bf*>(l) = *reinterpret_cast<const v8bf*>(g);
#endif
}

template <int N>
__device__ __forceinline__ void wait_async() {
#ifdef USE_ASYNC_LDS
  __builtin_amdgcn_s_wait_asynccnt(N);
#endif
}

__device__ __forceinline__ v8f vzero() {
  v8f v;
#pragma unroll
  for (int i = 0; i < 8; ++i) v[i] = 0.f;
  return v;
}

// Load a 16x32 bf16 operand fragment (A or B^T) from row-major [rows, ld]:
// lane 0-15 -> row rc+lane holds K = k0+{0..7,16..23};
// lane 16-31 -> row rc+(lane-16) holds K = k0+{8..15,24..31}.
__device__ __forceinline__ v16bf ldfrag(const bf16_t* __restrict__ base, int ld,
                                        int rc, int k0, int lane) {
  int r  = rc + (lane & 15);
  int kh = (lane >> 4) << 3;
  const bf16_t* p = base + (size_t)r * ld + (k0 + kh);
  union { v16bf v; v8bf h[2]; } u;
  u.h[0] = *reinterpret_cast<const v8bf*>(p);
  u.h[1] = *reinterpret_cast<const v8bf*>(p + 16);
  return u.v;
}

__device__ __forceinline__ v8f wmma_bf16(v16bf a, v16bf b, v8f c) {
  return __builtin_amdgcn_wmma_f32_16x16x32_bf16(false, a, false, b,
                                                 (short)0, c, false, false);
}

__device__ __forceinline__ float eluf(float v) { return v > 0.f ? v : expm1f(v); }
__device__ __forceinline__ float sigf(float v) { return 1.f / (1.f + expf(-v)); }

// ---------------------------------------------------------------- f32 -> bf16
__global__ void f32_to_bf16_k(const float* __restrict__ in, bf16_t* __restrict__ out, int n) {
  int i = blockIdx.x * 256 + threadIdx.x;
  if (i < n) out[i] = (bf16_t)in[i];
}

// ---------------------------------------------------------------- conv1: frame [TB,84,84,3] viewed as x[TB,3,84,84] (NCWH) -> bf16 [TB,32,42,42], ELU
__global__ __launch_bounds__(256) void conv1_k(const float* __restrict__ frame,
                                               const float* __restrict__ w,
                                               const float* __restrict__ b,
                                               bf16_t* __restrict__ out) {
  __shared__ float sw[864];   // [32][3][3][3]
  __shared__ float sb[32];
  int tid = threadIdx.x, img = blockIdx.x;
  for (int i = tid; i < 864; i += 256) sw[i] = w[i];
  if (tid < 32) sb[tid] = b[tid];
  __syncthreads();
  const float* fr = frame + (size_t)img * 84 * 84 * 3;   // [H][W][C]
  for (int p = tid; p < 42 * 42; p += 256) {
    int ow = p / 42, oh = p % 42;
    float acc[32];
#pragma unroll
    for (int oc = 0; oc < 32; ++oc) acc[oc] = sb[oc];
    for (int kw = 0; kw < 3; ++kw) {
      int iw = 2 * ow - 1 + kw; if (iw < 0 || iw >= 84) continue;
      for (int kh = 0; kh < 3; ++kh) {
        int ih = 2 * oh - 1 + kh; if (ih < 0 || ih >= 84) continue;
        for (int ci = 0; ci < 3; ++ci) {
          // x[c][iw][ih] == frame[ih][iw][c]
          float xv = fr[(ih * 84 + iw) * 3 + ci];
          int wi = ci * 9 + kw * 3 + kh;
#pragma unroll
          for (int oc = 0; oc < 32; ++oc) acc[oc] += sw[oc * 27 + wi] * xv;
        }
      }
    }
    bf16_t* op = out + (size_t)img * 32 * 42 * 42;
#pragma unroll
    for (int oc = 0; oc < 32; ++oc)
      op[(oc * 42 + ow) * 42 + oh] = (bf16_t)eluf(acc[oc]);
  }
}

// ---------------------------------------------------------------- conv2/3 (32->32 ch), ELU; flat!=0 stores [img][oc*121+ow*11+oh]
__global__ __launch_bounds__(256) void conv32_k(const bf16_t* __restrict__ in,
                                                const float* __restrict__ w,
                                                const float* __restrict__ b,
                                                bf16_t* __restrict__ out,
                                                int IN, int OUT, int flat) {
  __shared__ float sw[9216];  // [32][32][3][3]
  __shared__ float sb[32];
  int tid = threadIdx.x, img = blockIdx.x;
  for (int i = tid; i < 9216; i += 256) sw[i] = w[i];
  if (tid < 32) sb[tid] = b[tid];
  __syncthreads();
  const bf16_t* ip = in + (size_t)img * 32 * IN * IN;
  int npix = OUT * OUT;
  for (int p = tid; p < npix; p += 256) {
    int ow = p / OUT, oh = p % OUT;
    float acc[32];
#pragma unroll
    for (int oc = 0; oc < 32; ++oc) acc[oc] = sb[oc];
    for (int kw = 0; kw < 3; ++kw) {
      int iw = 2 * ow - 1 + kw; if (iw < 0 || iw >= IN) continue;
      for (int kh = 0; kh < 3; ++kh) {
        int ih = 2 * oh - 1 + kh; if (ih < 0 || ih >= IN) continue;
        for (int ci = 0; ci < 32; ++ci) {
          float xv = (float)ip[(ci * IN + iw) * IN + ih];
          int wi = ci * 9 + kw * 3 + kh;
#pragma unroll
          for (int oc = 0; oc < 32; ++oc) acc[oc] += sw[oc * 288 + wi] * xv;
        }
      }
    }
#pragma unroll
    for (int oc = 0; oc < 32; ++oc) {
      float v = eluf(acc[oc]);
      if (flat) out[(size_t)img * FLATK + oc * 121 + ow * 11 + oh] = (bf16_t)v;
      else      out[(((size_t)img * 32 + oc) * OUT + ow) * OUT + oh] = (bf16_t)v;
    }
  }
}

// ---------------------------------------------------------------- GEMM: out = relu(A[M,K] * B[N,K]^T + bias), bf16 in/out, f32 accum
// block: 256 thr = 8 waves in 4(row)x2(col) grid; tile 128 rows x 64 cols.
// A/B k-slabs double-buffered in LDS via async loads; grid (M/128, N/64).
__global__ __launch_bounds__(256) void gemm_bias_relu_k(const bf16_t* __restrict__ A,
                                                        const bf16_t* __restrict__ B,
                                                        const float* __restrict__ bias,
                                                        bf16_t* __restrict__ out,
                                                        int M, int N, int K) {
  __shared__ bf16_t sA[2][128 * 32];
  __shared__ bf16_t sB[2][64 * 32];
  const int tid  = threadIdx.x;
  const int lane = tid & 31;
  const int wave = tid >> 5;
  const int rw = wave & 3;       // row-wave 0..3 -> 32 rows each
  const int cw = wave >> 2;      // col-wave 0..1 -> 32 cols each
  const int rbase = blockIdx.x * 128;
  const int nbase = blockIdx.y * 64;
  const int KT = K >> 5;

  const int arow0 = tid >> 2;          // staging rows 0..63
  const int arow1 = arow0 + 64;        // staging rows 64..127
  const int akc   = (tid & 3) << 3;    // 8-elem chunk within 32-K slab

  auto stage = [&](int kt, int buf) {
    int k0 = kt << 5;
    copy16(A + (size_t)(rbase + arow0) * K + k0 + akc, &sA[buf][arow0 * 32 + akc]);
    copy16(A + (size_t)(rbase + arow1) * K + k0 + akc, &sA[buf][arow1 * 32 + akc]);
    copy16(B + (size_t)(nbase + arow0) * K + k0 + akc, &sB[buf][arow0 * 32 + akc]);
  };

  v8f acc[2][2];
#pragma unroll
  for (int rr = 0; rr < 2; ++rr)
#pragma unroll
    for (int cc = 0; cc < 2; ++cc) acc[rr][cc] = vzero();

  stage(0, 0);
  for (int kt = 0; kt < KT; ++kt) {
    int buf = kt & 1;
    if (kt + 1 < KT) { stage(kt + 1, buf ^ 1); wait_async<3>(); }
    else             { wait_async<0>(); }
    __syncthreads();
    v16bf a0 = ldfrag(&sA[buf][0], 32, rw * 32,      0, lane);
    v16bf a1 = ldfrag(&sA[buf][0], 32, rw * 32 + 16, 0, lane);
    v16bf b0 = ldfrag(&sB[buf][0], 32, cw * 32,      0, lane);
    v16bf b1 = ldfrag(&sB[buf][0], 32, cw * 32 + 16, 0, lane);
    acc[0][0] = wmma_bf16(a0, b0, acc[0][0]);
    acc[0][1] = wmma_bf16(a0, b1, acc[0][1]);
    acc[1][0] = wmma_bf16(a1, b0, acc[1][0]);
    acc[1][1] = wmma_bf16(a1, b1, acc[1][1]);
    __syncthreads();
  }
  int mhi = (lane >> 4) * 8;
#pragma unroll
  for (int rr = 0; rr < 2; ++rr) {
#pragma unroll
    for (int cc = 0; cc < 2; ++cc) {
      int n = nbase + cw * 32 + cc * 16 + (lane & 15);
      float bv = bias[n];
#pragma unroll
      for (int r = 0; r < 8; ++r) {
        int m = rbase + rw * 32 + rr * 16 + mhi + r;
        float v = acc[rr][cc][r] + bv;
        out[(size_t)m * N + n] = (bf16_t)(v > 0.f ? v : 0.f);
      }
    }
  }
}

// ---------------------------------------------------------------- LSTM gates: gates[32,4096] = X*Wih^T + Hh*Whh^T + (bih+bhh)
// block: 128 thr = 4 waves, each wave one 16-col group; grid = 4096/64.
// 64 k-slabs total (32 from X, 32 from Hh) double-buffered in LDS; weights stream from L2.
__global__ __launch_bounds__(128) void lstm_gates_k(const bf16_t* __restrict__ X,
                                                    const bf16_t* __restrict__ Wih,
                                                    const bf16_t* __restrict__ Hh,
                                                    const bf16_t* __restrict__ Whh,
                                                    const float* __restrict__ bih,
                                                    const float* __restrict__ bhh,
                                                    float* __restrict__ gates) {
  __shared__ bf16_t sX[2][32 * 32];
  const int tid  = threadIdx.x;
  const int lane = tid & 31;
  const int wave = tid >> 5;
  const int nbase = blockIdx.x * 64 + wave * 16;
  const int arow = tid >> 2;          // 0..31
  const int akc  = (tid & 3) << 3;

  auto stage = [&](int kt, int buf) {
    const bf16_t* src = (kt & 32) ? Hh : X;
    int k0 = (kt & 31) << 5;
    copy16(src + (size_t)arow * HID + k0 + akc, &sX[buf][arow * 32 + akc]);
  };

  v8f acc0 = vzero(), acc1 = vzero();
  stage(0, 0);
  for (int kt = 0; kt < 64; ++kt) {
    int buf = kt & 1;
    if (kt + 1 < 64) { stage(kt + 1, buf ^ 1); wait_async<1>(); }
    else             { wait_async<0>(); }
    __syncthreads();
    const bf16_t* W = (kt & 32) ? Whh : Wih;
    int k0 = (kt & 31) << 5;
    __builtin_prefetch(W + (size_t)(nbase + (lane & 15)) * HID + k0 + 256, 0, 1);
    v16bf b  = ldfrag(W, HID, nbase, k0, lane);
    v16bf a0 = ldfrag(&sX[buf][0], 32, 0,  0, lane);
    v16bf a1 = ldfrag(&sX[buf][0], 32, 16, 0, lane);
    acc0 = wmma_bf16(a0, b, acc0);
    acc1 = wmma_bf16(a1, b, acc1);
    __syncthreads();
  }
  int n = nbase + (lane & 15);
  float bv = bih[n] + bhh[n];
  int mhi = (lane >> 4) * 8;
#pragma unroll
  for (int r = 0; r < 8; ++r) {
    gates[(size_t)(mhi + r) * NG + n]      = acc0[r] + bv;
    gates[(size_t)(16 + mhi + r) * NG + n] = acc1[r] + bv;
  }
}

// ---------------------------------------------------------------- done-mask h,c and refresh bf16 h
__global__ void mask_hc_k(float* __restrict__ h, float* __restrict__ c,
                          bf16_t* __restrict__ hb, const int* __restrict__ done, int t) {
  int idx = blockIdx.x * blockDim.x + threadIdx.x;   // 2*32*1024
  int b = (idx >> 10) & 31;
  float nd = done[t * BB + b] ? 0.f : 1.f;
  float hv = h[idx] * nd;
  h[idx] = hv;
  c[idx] *= nd;
  hb[idx] = (bf16_t)hv;
}

// ---------------------------------------------------------------- LSTM pointwise cell update (one layer)
__global__ void lstm_cell_k(const float* __restrict__ gates, float* __restrict__ c,
                            float* __restrict__ h, bf16_t* __restrict__ hb,
                            float* __restrict__ coreout) {
  int idx = blockIdx.x * blockDim.x + threadIdx.x;   // 32*1024
  int b = idx >> 10, k = idx & 1023;
  const float* g = gates + (size_t)b * NG;
  float gi = g[k], gf = g[HID + k], gg = g[2 * HID + k], go = g[3 * HID + k];
  float cn = sigf(gf) * c[idx] + sigf(gi) * tanhf(gg);
  float hn = sigf(go) * tanhf(cn);
  c[idx] = cn;
  h[idx] = hn;
  hb[idx] = (bf16_t)hn;
  if (coreout) coreout[idx] = hn;
}

// ---------------------------------------------------------------- heads: logits[7], baseline, argmax per row; one wave per row
__global__ __launch_bounds__(32) void heads_k(const float* __restrict__ co,
                                              const float* __restrict__ pw,
                                              const float* __restrict__ pb,
                                              const float* __restrict__ bw,
                                              const float* __restrict__ bb,
                                              float* __restrict__ out) {
  int row = blockIdx.x, lane = threadIdx.x;
  float acc[8];
#pragma unroll
  for (int j = 0; j < 8; ++j) acc[j] = 0.f;
  for (int k = lane; k < HID; k += 32) {
    float xv = co[(size_t)row * HID + k];
#pragma unroll
    for (int j = 0; j < 7; ++j) acc[j] += xv * pw[j * HID + k];
    acc[7] += xv * bw[k];
  }
#pragma unroll
  for (int j = 0; j < 8; ++j)
    for (int off = 16; off; off >>= 1)
      acc[j] += __shfl_xor(acc[j], off, 32);
  if (lane == 0) {
    float best = -1e30f; int bi = 0;
#pragma unroll
    for (int j = 0; j < 7; ++j) {
      float v = acc[j] + pb[j];
      out[row * 7 + j] = v;
      if (v > best) { best = v; bi = j; }
    }
    out[TB * 7 + row] = acc[7] + bb[0];          // baseline at 7168
    out[TB * 7 + TB + row] = (float)bi;          // action at 8192
  }
}

// ================================================================ host
extern "C" void kernel_launch(void* const* d_in, const int* in_sizes, int n_in,
                              void* d_out, int out_size, void* d_ws, size_t ws_size,
                              hipStream_t stream) {
  const float* frame   = (const float*)d_in[0];
  const int*   done    = (const int*)  d_in[1];
  const float* h0      = (const float*)d_in[2];
  const float* c0      = (const float*)d_in[3];
  const float* conv_w1 = (const float*)d_in[4];
  const float* conv_b1 = (const float*)d_in[5];
  const float* conv_w2 = (const float*)d_in[6];
  const float* conv_b2 = (const float*)d_in[7];
  const float* conv_w3 = (const float*)d_in[8];
  const float* conv_b3 = (const float*)d_in[9];
  const float* fc_w1   = (const float*)d_in[10];
  const float* fc_b1   = (const float*)d_in[11];
  const float* fc_w2   = (const float*)d_in[12];
  const float* fc_b2   = (const float*)d_in[13];
  const float* wih0    = (const float*)d_in[14];
  const float* whh0    = (const float*)d_in[15];
  const float* bih0    = (const float*)d_in[16];
  const float* bhh0    = (const float*)d_in[17];
  const float* wih1    = (const float*)d_in[18];
  const float* whh1    = (const float*)d_in[19];
  const float* bih1    = (const float*)d_in[20];
  const float* bhh1    = (const float*)d_in[21];
  const float* pol_w   = (const float*)d_in[22];
  const float* pol_b   = (const float*)d_in[23];
  const float* base_w  = (const float*)d_in[24];
  const float* base_b  = (const float*)d_in[25];
  float* out = (float*)d_out;

  // -------- workspace carve-up
  char* ws = (char*)d_ws;
  size_t off = 0;
  auto take = [&](size_t bytes) -> char* {
    char* p = ws + off;
    off = (off + bytes + 255) & ~(size_t)255;
    return p;
  };
  bf16_t* c1o     = (bf16_t*)take((size_t)TB * 32 * 42 * 42 * 2);
  bf16_t* c2o     = (bf16_t*)take((size_t)TB * 32 * 21 * 21 * 2);
  bf16_t* flat    = (bf16_t*)take((size_t)TB * FLATK * 2);
  bf16_t* fc1o    = (bf16_t*)take((size_t)TB * HID * 2);
  bf16_t* corein  = (bf16_t*)take((size_t)TB * HID * 2);
  float*  coreout = (float*) take((size_t)TB * HID * 4);
  float*  hbuf    = (float*) take((size_t)2 * BB * HID * 4);
  float*  cbuf    = (float*) take((size_t)2 * BB * HID * 4);
  bf16_t* hbf     = (bf16_t*)take((size_t)2 * BB * HID * 2);
  float*  gates   = (float*) take((size_t)BB * NG * 4);
  bf16_t* bwfc1   = (bf16_t*)take((size_t)HID * FLATK * 2);
  bf16_t* bwfc2   = (bf16_t*)take((size_t)HID * HID * 2);
  bf16_t* bwih0   = (bf16_t*)take((size_t)NG * HID * 2);
  bf16_t* bwhh0   = (bf16_t*)take((size_t)NG * HID * 2);
  bf16_t* bwih1   = (bf16_t*)take((size_t)NG * HID * 2);
  bf16_t* bwhh1   = (bf16_t*)take((size_t)NG * HID * 2);

  auto cvt = [&](const float* s, bf16_t* d, int n) {
    f32_to_bf16_k<<<(n + 255) / 256, 256, 0, stream>>>(s, d, n);
  };
  cvt(fc_w1, bwfc1, HID * FLATK);
  cvt(fc_w2, bwfc2, HID * HID);
  cvt(wih0, bwih0, NG * HID);
  cvt(whh0, bwhh0, NG * HID);
  cvt(wih1, bwih1, NG * HID);
  cvt(whh1, bwhh1, NG * HID);

  // -------- conv stack (ELU), bf16 outputs
  conv1_k<<<TB, 256, 0, stream>>>(frame, conv_w1, conv_b1, c1o);
  conv32_k<<<TB, 256, 0, stream>>>(c1o, conv_w2, conv_b2, c2o, 42, 21, 0);
  conv32_k<<<TB, 256, 0, stream>>>(c2o, conv_w3, conv_b3, flat, 21, 11, 1);

  // -------- FC1 / FC2 (relu) via WMMA
  gemm_bias_relu_k<<<dim3(TB / 128, HID / 64), 256, 0, stream>>>(
      flat, bwfc1, fc_b1, fc1o, TB, HID, FLATK);
  gemm_bias_relu_k<<<dim3(TB / 128, HID / 64), 256, 0, stream>>>(
      fc1o, bwfc2, fc_b2, corein, TB, HID, HID);

  // -------- LSTM init state
  (void)hipMemcpyAsync(hbuf, h0, (size_t)2 * BB * HID * 4, hipMemcpyDeviceToDevice, stream);
  (void)hipMemcpyAsync(cbuf, c0, (size_t)2 * BB * HID * 4, hipMemcpyDeviceToDevice, stream);

  // -------- LSTM unroll over T
  for (int t = 0; t < TT; ++t) {
    mask_hc_k<<<(2 * BB * HID) / 256, 256, 0, stream>>>(hbuf, cbuf, hbf, done, t);
    // layer 0: x = corein[t], h = hbf[0]
    lstm_gates_k<<<NG / 64, 128, 0, stream>>>(
        corein + (size_t)t * BB * HID, bwih0, hbf, bwhh0, bih0, bhh0, gates);
    lstm_cell_k<<<(BB * HID) / 256, 256, 0, stream>>>(
        gates, cbuf, hbuf, hbf, (float*)nullptr);
    // layer 1: x = new h0 (hbf[0]), h = hbf[1]
    lstm_gates_k<<<NG / 64, 128, 0, stream>>>(
        hbf, bwih1, hbf + (size_t)BB * HID, bwhh1, bih1, bhh1, gates);
    lstm_cell_k<<<(BB * HID) / 256, 256, 0, stream>>>(
        gates, cbuf + (size_t)BB * HID, hbuf + (size_t)BB * HID,
        hbf + (size_t)BB * HID, coreout + (size_t)t * BB * HID);
  }

  // -------- heads: logits / baseline / argmax
  heads_k<<<TB, 32, 0, stream>>>(coreout, pol_w, pol_b, base_w, base_b, out);

  // -------- final state outputs: h_fin at 9216 floats, c_fin at 74752 floats
  (void)hipMemcpyAsync(out + 9216,  hbuf, (size_t)2 * BB * HID * 4, hipMemcpyDeviceToDevice, stream);
  (void)hipMemcpyAsync(out + 74752, cbuf, (size_t)2 * BB * HID * 4, hipMemcpyDeviceToDevice, stream);
}
